// LPLayerEncoder_42666205119207
// MI455X (gfx1250) — matrix operand order repaired and checked by reading, not compiled, not run on hardware
//
#include <hip/hip_runtime.h>
#include <hip/hip_bf16.h>
#include <math.h>

constexpr int cB = 4;
constexpr int cS = 4096;
constexpr int cN = 256;
constexpr int cD = 1024;
constexpr int cG = 16;
constexpr int cC = 64;
constexpr float cEPS = 1e-5f;

typedef __attribute__((ext_vector_type(16))) __bf16 bf16x16;
typedef __attribute__((ext_vector_type(8)))  __bf16 bf16x8;
typedef __attribute__((ext_vector_type(8)))  float  f32x8;

// ---------------- WMMA fragment loaders (bf16, contiguous k = 16*(lane/16) + s; A and B match) ----------------
// A 16x32 (MxK): lanes 0-15 rows M=0..15 with K 0..15; lanes 16-31 same rows, K 16..31
static __device__ inline bf16x16 frag_a_bf16(const __bf16* A, long lda, int m0, int k0, int lane) {
  int row = m0 + (lane & 15);
  int kb  = k0 + ((lane >> 4) << 4);
  const __bf16* p = A + (long)row * lda + kb;
  bf16x8 lo = *(const bf16x8*)(p);
  bf16x8 hi = *(const bf16x8*)(p + 8);
  return __builtin_shufflevector(lo, hi, 0,1,2,3,4,5,6,7,8,9,10,11,12,13,14,15);
}

static __device__ inline bf16x16 frag_a_f32(const float* A, long lda, int m0, int k0, int lane) {
  int row = m0 + (lane & 15);
  int kb  = k0 + ((lane >> 4) << 4);
  const float* p = A + (long)row * lda + kb;
  bf16x16 a;
  #pragma unroll
  for (int j = 0; j < 8; ++j) a[j]     = (__bf16)p[j];
  #pragma unroll
  for (int j = 0; j < 8; ++j) a[8 + j] = (__bf16)p[8 + j];
  return a;
}

// B 32x16 (KxN), element (k,n) stored at Bw[n*ldb + k]  ("weight" layout: y = x @ W^T)
// lanes 0-15: col N=lane, K=0..15 ; lanes 16-31: col N=lane-16, K=16..31 -> contiguous 32B
static __device__ inline bf16x16 frag_b_kcontig(const __bf16* Bw, long ldb, int k0, int n0, int lane) {
  int col = n0 + (lane & 15);
  int kb  = k0 + ((lane >> 4) << 4);
  const __bf16* p = Bw + (long)col * ldb + kb;
  return *(const bf16x16*)p;
}

// B 32x16, element (k,n) stored at Bm[k*ldb + n] (row-major over n -> strided gathers)
static __device__ inline bf16x16 frag_b_strided(const __bf16* Bm, long ldb, int k0, int n0, int lane) {
  int col = n0 + (lane & 15);
  int kb  = k0 + ((lane >> 4) << 4);
  bf16x16 b;
  #pragma unroll
  for (int j = 0; j < 16; ++j) b[j] = Bm[(long)(kb + j) * ldb + col];
  return b;
}

// ---------------- generic WMMA GEMM: block 256 thr = 8 waves, tile 128x64 ----------------
enum { EPI_BF16 = 0, EPI_F32_SCALE = 1, EPI_BIAS_RES_F32 = 2, EPI_RES_F32 = 3, EPI_BIAS_GELU_BF16 = 4 };

template<bool AF32, bool BSTRID, int EPI>
__global__ __launch_bounds__(256) void gemm_k(
    const void* __restrict__ Ap, long lda, long aOffB, long aOffG,
    const __bf16* __restrict__ Bw, long ldb, long bOffB, long bOffG,
    void* __restrict__ Cp, long ldc, long cOffB, long cOffG,
    const float* __restrict__ bias,
    const float* __restrict__ resid, long rOffB,
    int K, int zdiv, float scale)
{
  int z  = blockIdx.z;
  int zb = z / zdiv;
  int zg = z % zdiv;
  long aoff = (long)zb * aOffB + (long)zg * aOffG;
  long boff = (long)zb * bOffB + (long)zg * bOffG;
  long coff = (long)zb * cOffB + (long)zg * cOffG;

  int lane = threadIdx.x & 31;
  int wave = threadIdx.x >> 5;
  int m0 = blockIdx.x * 128 + wave * 16;
  int n0 = blockIdx.y * 64;

  const __bf16* Ab = (const __bf16*)Ap + aoff;
  const float*  Af = (const float*)Ap  + aoff;
  const __bf16* Bb = Bw + boff;

  f32x8 acc[4] = {};

  for (int k0 = 0; k0 < K; k0 += 32) {
    if (k0 + 32 < K) {  // gfx1250 global_prefetch_b8 path
      const char* pf = AF32 ? (const char*)(Af + (long)(m0 + (lane & 15)) * lda + k0 + 32)
                            : (const char*)(Ab + (long)(m0 + (lane & 15)) * lda + k0 + 32);
      __builtin_prefetch(pf, 0, 0);
    }
    bf16x16 a = AF32 ? frag_a_f32(Af, lda, m0, k0, lane)
                     : frag_a_bf16(Ab, lda, m0, k0, lane);
    #pragma unroll
    for (int t = 0; t < 4; ++t) {
      bf16x16 b = BSTRID ? frag_b_strided(Bb, ldb, k0, n0 + t * 16, lane)
                         : frag_b_kcontig(Bb, ldb, k0, n0 + t * 16, lane);
      acc[t] = __builtin_amdgcn_wmma_f32_16x16x32_bf16(
          false, a, false, b, (short)0, acc[t], false, false);
    }
  }

  // D layout: lane l, vgpr r -> m = m0 + r + (l>=16 ? 8 : 0), n = n0 + (l&15)
  #pragma unroll
  for (int t = 0; t < 4; ++t) {
    #pragma unroll
    for (int r = 0; r < 8; ++r) {
      int m = m0 + r + ((lane >> 4) << 3);
      int n = n0 + t * 16 + (lane & 15);
      float v = acc[t][r] * scale;
      long ci = coff + (long)m * ldc + n;
      if (EPI == EPI_BF16) {
        ((__bf16*)Cp)[ci] = (__bf16)v;
      } else if (EPI == EPI_F32_SCALE) {
        ((float*)Cp)[ci] = v;
      } else if (EPI == EPI_BIAS_RES_F32) {
        v += bias[n] + resid[(long)zb * rOffB + (long)m * ldc + n];
        ((float*)Cp)[ci] = v;
      } else if (EPI == EPI_RES_F32) {
        v += resid[(long)zb * rOffB + (long)m * ldc + n];
        ((float*)Cp)[ci] = v;
      } else { // EPI_BIAS_GELU_BF16 : exact gelu
        v += bias[n];
        float gl = 0.5f * v * (1.0f + erff(v * 0.70710678118654752f));
        ((__bf16*)Cp)[ci] = (__bf16)gl;
      }
    }
  }
}

// ---------------- LayerNorm over D=1024, one block (256 thr) per row, bf16 out ----------------
// mode 0: x = in0 ; mode 1: x = in0 + in1 (same shape) ; mode 2: x = in0 + in1[row*cC + (d%cC)]
__global__ __launch_bounds__(256) void ln_k(
    const float* __restrict__ in0, const float* __restrict__ in1, int mode,
    const float* __restrict__ w, const float* __restrict__ bb, __bf16* __restrict__ out)
{
  __shared__ float red[256];
  int tid = threadIdx.x;
  long row = blockIdx.x;
  long base = row * cD;
  float x[4];
  #pragma unroll
  for (int i = 0; i < 4; ++i) {
    int d = tid + i * 256;
    float v = in0[base + d];
    if (mode == 1)      v += in1[base + d];
    else if (mode == 2) v += in1[row * cC + (d & (cC - 1))];
    x[i] = v;
  }
  float s = x[0] + x[1] + x[2] + x[3];
  red[tid] = s; __syncthreads();
  for (int off = 128; off > 0; off >>= 1) { if (tid < off) red[tid] += red[tid + off]; __syncthreads(); }
  float mu = red[0] * (1.0f / cD);
  __syncthreads();
  float vs = 0.f;
  #pragma unroll
  for (int i = 0; i < 4; ++i) { float dd = x[i] - mu; vs += dd * dd; }
  red[tid] = vs; __syncthreads();
  for (int off = 128; off > 0; off >>= 1) { if (tid < off) red[tid] += red[tid + off]; __syncthreads(); }
  float rstd = rsqrtf(red[0] * (1.0f / cD) + cEPS);
  #pragma unroll
  for (int i = 0; i < 4; ++i) {
    int d = tid + i * 256;
    out[base + d] = (__bf16)((x[i] - mu) * rstd * w[d] + bb[d]);
  }
}

// ---------------- masked softmax over S=4096; one block per (b,n,g) row, in place ----------------
__global__ __launch_bounds__(256) void softmax_k(float* __restrict__ attn,
                                                 const unsigned char* __restrict__ mask)
{
  __shared__ float red[256];
  int tid = threadIdx.x;
  long r = blockIdx.x;                 // enumerates (b,n,g) row-major
  int b = (int)(r / ((long)cN * cG));
  float* p = attn + r * cS;
  const unsigned char* mrow = mask + (long)b * cS;
  float x[16];
  float mx = -3.0e38f;
  #pragma unroll
  for (int i = 0; i < 16; ++i) {
    int s = tid + i * 256;
    float v = mrow[s] ? -3.0e38f : p[s];
    x[i] = v;
    mx = fmaxf(mx, v);
  }
  red[tid] = mx; __syncthreads();
  for (int off = 128; off > 0; off >>= 1) { if (tid < off) red[tid] = fmaxf(red[tid], red[tid + off]); __syncthreads(); }
  mx = red[0]; __syncthreads();
  float sm = 0.f;
  #pragma unroll
  for (int i = 0; i < 16; ++i) {
    float e = (x[i] <= -1.0e37f) ? 0.f : __expf(x[i] - mx);
    x[i] = e; sm += e;
  }
  red[tid] = sm; __syncthreads();
  for (int off = 128; off > 0; off >>= 1) { if (tid < off) red[tid] += red[tid + off]; __syncthreads(); }
  float inv = 1.0f / red[0];
  #pragma unroll
  for (int i = 0; i < 16; ++i) p[tid + i * 256] = x[i] * inv;
}

// ---------------- V transpose: vT[b][g][c][s] = v[b][s][g*cC+c] (bf16) ----------------
__global__ __launch_bounds__(256) void vtrans_k(const __bf16* __restrict__ vb, __bf16* __restrict__ vT)
{
  long i = (long)blockIdx.x * 256 + threadIdx.x;      // over B*G*C*S
  int s = (int)(i & (cS - 1));
  long t = i >> 12;                                   // / cS
  int c = (int)(t & (cC - 1));
  long t2 = t >> 6;                                   // / cC
  int g = (int)(t2 & (cG - 1));
  int b = (int)(t2 >> 4);                             // / cG
  vT[i] = vb[((long)(b * cS + s)) * cD + g * cC + c];
}

// ---------------- f32 -> bf16 convert ----------------
__global__ __launch_bounds__(256) void cvt_k(const float* __restrict__ in, __bf16* __restrict__ out, int n)
{
  int i = blockIdx.x * 256 + threadIdx.x;
  if (i < n) out[i] = (__bf16)in[i];
}

extern "C" void kernel_launch(void* const* d_in, const int* in_sizes, int n_in,
                              void* d_out, int out_size, void* d_ws, size_t ws_size,
                              hipStream_t stream)
{
  (void)in_sizes; (void)n_in; (void)out_size; (void)ws_size;
  const float* feats  = (const float*)d_in[0];
  const float* parts  = (const float*)d_in[1];
  const float* qpos   = (const float*)d_in[2];
  const float* kpos   = (const float*)d_in[3];
  const unsigned char* mask = (const unsigned char*)d_in[4];
  const float* ln_q_w = (const float*)d_in[5];
  const float* ln_q_b = (const float*)d_in[6];
  const float* ln_k_w = (const float*)d_in[7];
  const float* ln_k_b = (const float*)d_in[8];
  const float* ln_v_w = (const float*)d_in[9];
  const float* ln_v_b = (const float*)d_in[10];
  const float* wq     = (const float*)d_in[11];
  const float* wk     = (const float*)d_in[12];
  const float* wv     = (const float*)d_in[13];
  const float* w_proj = (const float*)d_in[14];
  const float* b_proj = (const float*)d_in[15];
  const float* r_ln_w = (const float*)d_in[16];
  const float* r_ln_b = (const float*)d_in[17];
  const float* reason = (const float*)d_in[18];
  const float* m_ln_w = (const float*)d_in[19];
  const float* m_ln_b = (const float*)d_in[20];
  const float* fc1_w  = (const float*)d_in[21];
  const float* fc1_b  = (const float*)d_in[22];
  const float* fc2_w  = (const float*)d_in[23];
  const float* fc2_b  = (const float*)d_in[24];

  const size_t BN = (size_t)cB * cN;
  const size_t BS = (size_t)cB * cS;

  char* wp = (char*)d_ws;
  auto carve = [&](size_t bytes) -> char* {
    char* p = wp; wp += (bytes + 255) & ~(size_t)255; return p;
  };
  __bf16* qin  = (__bf16*)carve(BN * cD * 2);
  __bf16* kin  = (__bf16*)carve(BS * cD * 2);
  __bf16* vin  = (__bf16*)carve(BS * cD * 2);
  __bf16* qb   = (__bf16*)carve(BN * cD * 2);
  __bf16* kb   = (__bf16*)carve(BS * cD * 2);
  __bf16* vb   = (__bf16*)carve(BS * cD * 2);
  __bf16* vT   = kin;   // kin dead after K projection -> reuse 32MB for transposed V
  __bf16* aob  = (__bf16*)carve(BN * cD * 2);     // attention output (B,N,D) bf16
  float*  p1   = (float*) carve(BN * cD * 4);     // parts after attn residual
  float*  p2   = (float*) carve(BN * cD * 4);     // parts after reason residual
  __bf16* tokb = (__bf16*)carve(BN * cD * 2);
  __bf16* mlpb = (__bf16*)carve(BN * cD * 2);
  __bf16* hb   = (__bf16*)carve(BN * cD * 2);
  __bf16* wqb  = (__bf16*)carve((size_t)cD * cD * 2);
  __bf16* wkb  = (__bf16*)carve((size_t)cD * cD * 2);
  __bf16* wvb  = (__bf16*)carve((size_t)cD * cD * 2);
  __bf16* wpb  = (__bf16*)carve((size_t)cD * cD * 2);
  __bf16* f1b  = (__bf16*)carve((size_t)cD * cD * 2);
  __bf16* f2b  = (__bf16*)carve((size_t)cD * cD * 2);
  __bf16* rwb  = (__bf16*)carve((size_t)cN * cN * 2);

  float* outparts = (float*)d_out;                       // (B,N,D)
  float* attn     = (float*)d_out + BN * cD;             // (B,N,G,S)

  // ---- weights to bf16 ----
  const int nw = cD * cD;
  cvt_k<<<nw / 256, 256, 0, stream>>>(wq,     wqb, nw);
  cvt_k<<<nw / 256, 256, 0, stream>>>(wk,     wkb, nw);
  cvt_k<<<nw / 256, 256, 0, stream>>>(wv,     wvb, nw);
  cvt_k<<<nw / 256, 256, 0, stream>>>(w_proj, wpb, nw);
  cvt_k<<<nw / 256, 256, 0, stream>>>(fc1_w,  f1b, nw);
  cvt_k<<<nw / 256, 256, 0, stream>>>(fc2_w,  f2b, nw);
  cvt_k<<<(cN * cN) / 256, 256, 0, stream>>>(reason, rwb, cN * cN);

  // ---- LN inputs (fused adds) ----
  ln_k<<<(int)BN, 256, 0, stream>>>(parts, qpos, 2, ln_q_w, ln_q_b, qin);
  ln_k<<<(int)BS, 256, 0, stream>>>(feats, kpos, 1, ln_k_w, ln_k_b, kin);
  ln_k<<<(int)BS, 256, 0, stream>>>(feats, nullptr, 0, ln_v_w, ln_v_b, vin);

  // ---- Q/K/V projections (y = x @ W^T), bf16 out ----
  gemm_k<false, false, EPI_BF16><<<dim3(8, 16, 1), 256, 0, stream>>>(
      qin, cD, 0, 0, wqb, cD, 0, 0, qb, cD, 0, 0, nullptr, nullptr, 0, cD, 1, 1.0f);
  gemm_k<false, false, EPI_BF16><<<dim3(128, 16, 1), 256, 0, stream>>>(
      kin, cD, 0, 0, wkb, cD, 0, 0, kb, cD, 0, 0, nullptr, nullptr, 0, cD, 1, 1.0f);
  gemm_k<false, false, EPI_BF16><<<dim3(128, 16, 1), 256, 0, stream>>>(
      vin, cD, 0, 0, wvb, cD, 0, 0, vb, cD, 0, 0, nullptr, nullptr, 0, cD, 1, 1.0f);

  // ---- V -> [b][g][c][s] so attn@V B-fragments are K-contiguous ----
  vtrans_k<<<(int)(((long)cB * cG * cC * cS) / 256), 256, 0, stream>>>(vb, vT);

  // ---- scores = scale * Q K^T per (b,g), written straight into d_out attn region ----
  gemm_k<false, false, EPI_F32_SCALE><<<dim3(2, 64, 64), 256, 0, stream>>>(
      qb,   cD,            (long)cN * cD,       (long)cC,
      kb,   cD,            (long)cS * cD,       (long)cC,
      attn, (long)cG * cS, (long)cN * cG * cS,  (long)cS,
      nullptr, nullptr, 0, cC, 16, 0.125f);

  // ---- masked softmax in place ----
  softmax_k<<<cB * cN * cG, 256, 0, stream>>>(attn, mask);

  // ---- out = attn @ V per (b,g) -> (B,N,D) bf16 ----
  gemm_k<true, false, EPI_BF16><<<dim3(2, 1, 64), 256, 0, stream>>>(
      attn, (long)cG * cS, (long)cN * cG * cS, (long)cS,
      vT,   (long)cS,      (long)cG * cC * cS, (long)cC * cS,
      aob,  (long)cD,      (long)cN * cD,      (long)cC,
      nullptr, nullptr, 0, cS, 16, 1.0f);

  // ---- parts1 = parts + out @ Wp^T + b_proj ----
  gemm_k<false, false, EPI_BIAS_RES_F32><<<dim3(8, 16, 1), 256, 0, stream>>>(
      aob, cD, 0, 0, wpb, cD, 0, 0, p1, cD, 0, 0, b_proj, parts, 0, cD, 1, 1.0f);

  // ---- reason mix: parts2 = parts1 + reason_w @ LN(parts1) ----
  ln_k<<<(int)BN, 256, 0, stream>>>(p1, nullptr, 0, r_ln_w, r_ln_b, tokb);
  gemm_k<false, true, EPI_RES_F32><<<dim3(2, 16, 4), 256, 0, stream>>>(
      rwb,  (long)cN, 0, 0,
      tokb, (long)cD, (long)cN * cD, 0,
      p2,   (long)cD, (long)cN * cD, 0,
      nullptr, p1, (long)cN * cD, cN, 1, 1.0f);

  // ---- MLP: parts_out = parts2 + gelu(LN(parts2)@fc1^T+b1)@fc2^T + b2 ----
  ln_k<<<(int)BN, 256, 0, stream>>>(p2, nullptr, 0, m_ln_w, m_ln_b, mlpb);
  gemm_k<false, false, EPI_BIAS_GELU_BF16><<<dim3(8, 16, 1), 256, 0, stream>>>(
      mlpb, cD, 0, 0, f1b, cD, 0, 0, hb, cD, 0, 0, fc1_b, nullptr, 0, cD, 1, 1.0f);
  gemm_k<false, false, EPI_BIAS_RES_F32><<<dim3(8, 16, 1), 256, 0, stream>>>(
      hb, cD, 0, 0, f2b, cD, 0, 0, outparts, cD, 0, 0, fc2_b, p2, 0, cD, 1, 1.0f);
}
